// GroupQueryFlashAttention_46712064312076
// MI455X (gfx1250) — matrix operand order repaired and checked by reading, not compiled, not run on hardware
//
#include <hip/hip_runtime.h>
#include <hip/hip_bf16.h>

// ---------------- problem constants ----------------
#define BB   2
#define SS   2048
#define DM   1024
#define HH   16
#define HKV  4
#define HD   64
#define REP  (HH / HKV)
#define NQKV (HH * HD + 2 * HKV * HD)   // 1536
#define ROWS (BB * SS)                  // 4096

typedef __attribute__((ext_vector_type(16))) _Float16 v16h;
typedef __attribute__((ext_vector_type(8)))  _Float16 v8h;
typedef __attribute__((ext_vector_type(8)))  float    v8f;

// D = A(16x32 f16) * B(32x16 f16) + C(16x16 f32)
__device__ __forceinline__ v8f wmma_f16(v16h a, v16h b, v8f c) {
  return __builtin_amdgcn_wmma_f32_16x16x32_f16(
      false, a, false, b, (short)0, c, false, false);
}

// ---- A fragment (16x32, 16-bit): lane l holds row m=l&15; halves j=0..15 map
//      to k = (j<8 ? (l>>4)*8 + j : 16 + (l>>4)*8 + (j-8))  (ISA 7.12.2) ----
__device__ __forceinline__ v16h load_a_frag(const _Float16* A, int lda,
                                            int row_base, int k_base) {
  int lane = threadIdx.x & 31;
  int m = lane & 15, hi = lane >> 4;
  const _Float16* p = A + (size_t)(row_base + m) * lda + k_base + hi * 8;
  v8h lo = *(const v8h*)p;          // k = hi*8 .. hi*8+7
  v8h hg = *(const v8h*)(p + 16);   // k = 16+hi*8 .. 16+hi*8+7
  v16h r;
#pragma unroll
  for (int i = 0; i < 8; ++i) { r[i] = lo[i]; r[i + 8] = hg[i]; }
  return r;
}

// ---- B fragment (32x16, 16-bit): lane l holds col n=l&15; halves j=0..15 map
//      to k = (j<8 ? (l>>4)*8 + j : 16 + (l>>4)*8 + (j-8)).  Bt is N-major (row n = one K-column), so each
//      lane reads two 16-byte chunks. ----
__device__ __forceinline__ v16h load_b_frag(const _Float16* Bt, int ldb,
                                            int n_base, int k_base) {
  int lane = threadIdx.x & 31;
  int n = lane & 15, hi = lane >> 4;
  const _Float16* p = Bt + (size_t)(n_base + n) * ldb + k_base + hi * 8;
  v8h a = *(const v8h*)p;
  v8h b = *(const v8h*)(p + 16);
  v16h r;
#pragma unroll
  for (int i = 0; i < 8; ++i) { r[i] = a[i]; r[i + 8] = b[i]; }
  return r;
}

// ---------------- conversion / packing kernels ----------------
__global__ void k_cvt_x(const float* __restrict__ x, _Float16* __restrict__ o, int n) {
  int i = blockIdx.x * blockDim.x + threadIdx.x;
  if (i < n) o[i] = (_Float16)x[i];
}

__global__ void k_pack_wqkv(const float* __restrict__ Wq, const float* __restrict__ Wk,
                            const float* __restrict__ Wv, _Float16* __restrict__ wt) {
  int i = blockIdx.x * blockDim.x + threadIdx.x;
  if (i >= NQKV * DM) return;
  int n = i / DM, k = i % DM;
  float v;
  if (n < HH * HD)                 v = Wq[(size_t)k * (HH * HD) + n];
  else if (n < HH * HD + HKV * HD) v = Wk[(size_t)k * (HKV * HD) + (n - HH * HD)];
  else                             v = Wv[(size_t)k * (HKV * HD) + (n - HH * HD - HKV * HD)];
  wt[(size_t)n * DM + k] = (_Float16)v;
}

__global__ void k_pack_wo(const float* __restrict__ Wo, _Float16* __restrict__ wt) {
  int i = blockIdx.x * blockDim.x + threadIdx.x;
  if (i >= DM * DM) return;
  int n = i / DM, k = i % DM;
  wt[(size_t)n * DM + k] = (_Float16)Wo[(size_t)k * DM + n];
}

// ---------------- LDS-tiled WMMA GEMM: C(MxN) = A(MxK) * Bt(NxK)^T ----------
// 256 threads = 8 waves; block tile 128(M) x 64(N); K-step 32.
// Wave w owns rows w*16..w*16+15 and all 64 columns (4 C fragments).
template <bool F16OUT>
__global__ __launch_bounds__(256, 1)
void k_gemm(const _Float16* __restrict__ A, const _Float16* __restrict__ Bt,
            void* __restrict__ C, int M, int N, int K) {
  __shared__ __align__(16) _Float16 As[128][32];
  __shared__ __align__(16) _Float16 Bs[64][32];
  int wave = threadIdx.x >> 5;
  int lane = threadIdx.x & 31;
  int m0 = blockIdx.y * 128;
  int n0 = blockIdx.x * 64;

  v8f c0 = {}, c1 = {}, c2 = {}, c3 = {};
  for (int kb = 0; kb < K; kb += 32) {
    __syncthreads();
    // stage A tile: 128x32 halves = 512 16B chunks, 2 per thread
#pragma unroll
    for (int c = 0; c < 2; ++c) {
      int t = threadIdx.x + c * 256;
      int r = t >> 2, kg = t & 3;
      *(v8h*)&As[r][kg * 8] = *(const v8h*)(A + (size_t)(m0 + r) * K + kb + kg * 8);
    }
    // stage B tile: 64x32 halves = 256 16B chunks, 1 per thread
    {
      int r = threadIdx.x >> 2, kg = threadIdx.x & 3;
      *(v8h*)&Bs[r][kg * 8] = *(const v8h*)(Bt + (size_t)(n0 + r) * K + kb + kg * 8);
    }
    if (kb + 32 < K)
      __builtin_prefetch(A + (size_t)(m0 + (threadIdx.x >> 1)) * K + kb + 32, 0, 0);
    __syncthreads();

    v16h a  = load_a_frag(&As[0][0], 32, wave * 16, 0);
    v16h b0 = load_b_frag(&Bs[0][0], 32,  0, 0);
    v16h b1 = load_b_frag(&Bs[0][0], 32, 16, 0);
    c0 = wmma_f16(a, b0, c0);
    c1 = wmma_f16(a, b1, c1);
    v16h b2 = load_b_frag(&Bs[0][0], 32, 32, 0);
    v16h b3 = load_b_frag(&Bs[0][0], 32, 48, 0);
    c2 = wmma_f16(a, b2, c2);
    c3 = wmma_f16(a, b3, c3);
  }

  int n = lane & 15, mlo = (lane >> 4) * 8;
#pragma unroll
  for (int v = 0; v < 8; ++v) {
    size_t row = (size_t)(m0 + wave * 16 + mlo + v);
    if (F16OUT) {
      _Float16* cr = (_Float16*)C + row * N + n0 + n;
      cr[0]  = (_Float16)c0[v];
      cr[16] = (_Float16)c1[v];
      cr[32] = (_Float16)c2[v];
      cr[48] = (_Float16)c3[v];
    } else {
      float* cr = (float*)C + row * N + n0 + n;
      cr[0]  = c0[v];
      cr[16] = c1[v];
      cr[32] = c2[v];
      cr[48] = c3[v];
    }
  }
}

// ---------------- 2D continuous RoPE, in place on q (cols 0..1023) and k (1024..1279)
__global__ void k_rope(_Float16* __restrict__ qkv, const float* __restrict__ pos) {
  const int SLOTS = HH + HKV;                 // 16 q heads + 4 kv heads
  int idx = blockIdx.x * blockDim.x + threadIdx.x;
  if (idx >= ROWS * SLOTS * 32) return;
  int p    = idx & 31;           // pair index within head
  int t    = idx >> 5;
  int slot = t % SLOTS;
  int row  = t / SLOTS;
  int s    = row % SS;
  int axis = p >> 4;             // 0 = x-half, 1 = y-half
  int j    = p & 15;             // frequency index within half
  int col  = (slot < HH ? slot * HD : DM + (slot - HH) * HD) + axis * 32 + 2 * j;
  float c  = pos[s * 2 + axis];
  float f  = c * __powf(10000.0f, -(float)j / 16.0f);   // inv_freq = theta^(-j/16)
  float sn, cs;
  __sincosf(f, &sn, &cs);
  _Float16* ptr = qkv + (size_t)row * NQKV + col;
  float x1 = (float)ptr[0], x2 = (float)ptr[1];
  ptr[0] = (_Float16)(x1 * cs - x2 * sn);
  ptr[1] = (_Float16)(x2 * cs + x1 * sn);
}

// ---------------- flash attention ----------------
// grid: (S/64, B*H); 128 threads = 4 waves; wave owns 16 query rows.
// K and V chunks (32 keys) staged once per block in LDS; online softmax with
// shuffle row-max; row-sum computed by a 5th WMMA against an all-ones B frag.
// __launch_bounds__(128, 1): allow the full VGPR budget so Q fragments and the
// five f32 accumulators stay resident (no scratch spills); latency is hidden
// by the 1024-block grid across WGPs, not by per-SIMD multi-wave occupancy.
__global__ __launch_bounds__(128, 1)
void k_attn(const _Float16* __restrict__ qkv, _Float16* __restrict__ out) {
  __shared__ __align__(16) _Float16 Ks[32][HD];        // K chunk, row-major (key, d)
  __shared__ __align__(16) _Float16 Vt[HD][32];        // V chunk, transposed (d, key)
  __shared__ __align__(16) _Float16 Plds[4][16][32];   // per-wave P round-trip

  int bh = blockIdx.y;
  int b = bh / HH, h = bh % HH, hkv = h / REP;
  int wave = threadIdx.x >> 5, lane = threadIdx.x & 31;
  int qrow0 = blockIdx.x * 64 + wave * 16;

  const _Float16* qbase = qkv + (size_t)b * SS * NQKV + h * HD;
  const _Float16* kbase = qkv + (size_t)b * SS * NQKV + DM + hkv * HD;
  const _Float16* vbase = qkv + (size_t)b * SS * NQKV + DM + HKV * HD + hkv * HD;

  // Q fragments, pre-scaled by 1/sqrt(HD) = 0.125
  v16h a0 = load_a_frag(qbase, NQKV, qrow0, 0);
  v16h a1 = load_a_frag(qbase, NQKV, qrow0, 32);
  v16h ones;
#pragma unroll
  for (int i = 0; i < 16; ++i) {
    a0[i] = a0[i] * (_Float16)0.125f;
    a1[i] = a1[i] * (_Float16)0.125f;
    ones[i] = (_Float16)1.0f;
  }

  v8f o0 = {}, o1 = {}, o2 = {}, o3 = {}, o4 = {};   // o4 = running row sums
  float rmax[8];
#pragma unroll
  for (int v = 0; v < 8; ++v) rmax[v] = -1e30f;

  int n  = lane & 15;
  int m0 = (lane >> 4) * 8;  // C-frag: vgpr v holds row m0+v

  for (int kc = 0; kc < SS; kc += 32) {
    __syncthreads();
    // stage K chunk row-major: 32x64 halves = 256 16B chunks, 2 per thread
#pragma unroll
    for (int c = 0; c < 2; ++c) {
      int t = threadIdx.x + c * 128;
      int key = t >> 3, kg = t & 7;
      *(v8h*)&Ks[key][kg * 8] =
          *(const v8h*)(kbase + (size_t)(kc + key) * NQKV + kg * 8);
    }
    // stage V chunk transposed: scalar stores
    for (int t = threadIdx.x; t < 32 * HD; t += 128) {
      int key = t >> 6, d = t & 63;
      Vt[d][key] = vbase[(size_t)(kc + key) * NQKV + d];
    }
    __syncthreads();

    // scores: two 16x16 tiles over 32 keys
    v8f z = {};
    v16h bk0 = load_b_frag(&Ks[0][0], HD, 0,  0);
    v16h bk1 = load_b_frag(&Ks[0][0], HD, 0, 32);
    v8f s0 = wmma_f16(a1, bk1, wmma_f16(a0, bk0, z));
    v16h bk2 = load_b_frag(&Ks[0][0], HD, 16,  0);
    v16h bk3 = load_b_frag(&Ks[0][0], HD, 16, 32);
    v8f s1 = wmma_f16(a1, bk3, wmma_f16(a0, bk2, z));

    // online softmax: shuffle row-max, write P directly to LDS, rescale accs
#pragma unroll
    for (int v = 0; v < 8; ++v) {
      float mv = fmaxf(s0[v], s1[v]);
      mv = fmaxf(mv, __shfl_xor(mv, 1, 32));
      mv = fmaxf(mv, __shfl_xor(mv, 2, 32));
      mv = fmaxf(mv, __shfl_xor(mv, 4, 32));
      mv = fmaxf(mv, __shfl_xor(mv, 8, 32));
      float nm = fmaxf(rmax[v], mv);
      float sc = __expf(rmax[v] - nm);          // exp(-inf)=0 on first chunk
      rmax[v] = nm;
      o0[v] *= sc; o1[v] *= sc; o2[v] *= sc; o3[v] *= sc; o4[v] *= sc;
      Plds[wave][m0 + v][n]      = (_Float16)__expf(s0[v] - nm);
      Plds[wave][m0 + v][n + 16] = (_Float16)__expf(s1[v] - nm);
    }
    asm volatile("s_wait_dscnt 0" ::: "memory");
    v16h ap = load_a_frag(&Plds[wave][0][0], 32, 0, 0);

    // O += P(16x32) . V_chunk(32x64); 5th WMMA accumulates row sums
    v16h vb0 = load_b_frag(&Vt[0][0], 32,  0, 0);
    v16h vb1 = load_b_frag(&Vt[0][0], 32, 16, 0);
    o0 = wmma_f16(ap, vb0, o0);
    o1 = wmma_f16(ap, vb1, o1);
    v16h vb2 = load_b_frag(&Vt[0][0], 32, 32, 0);
    v16h vb3 = load_b_frag(&Vt[0][0], 32, 48, 0);
    o2 = wmma_f16(ap, vb2, o2);
    o3 = wmma_f16(ap, vb3, o3);
    o4 = wmma_f16(ap, ones, o4);
  }

  // normalize and store (B,S,H*HD) fp16, head-major columns
#pragma unroll
  for (int v = 0; v < 8; ++v) {
    float inv = 1.0f / o4[v];                  // every column of o4 = row sum
    size_t row = (size_t)(b * SS + blockIdx.x * 64 + wave * 16 + m0 + v);
    _Float16* orow = out + row * (HH * HD) + h * HD;
    orow[n]      = (_Float16)(o0[v] * inv);
    orow[16 + n] = (_Float16)(o1[v] * inv);
    orow[32 + n] = (_Float16)(o2[v] * inv);
    orow[48 + n] = (_Float16)(o3[v] * inv);
  }
}

// ---------------- launch ----------------
extern "C" void kernel_launch(void* const* d_in, const int* in_sizes, int n_in,
                              void* d_out, int out_size, void* d_ws, size_t ws_size,
                              hipStream_t stream) {
  const float* x   = (const float*)d_in[0];
  const float* pos = (const float*)d_in[1];
  const float* Wq  = (const float*)d_in[2];
  const float* Wk  = (const float*)d_in[3];
  const float* Wv  = (const float*)d_in[4];
  const float* Wo  = (const float*)d_in[5];
  float* out = (float*)d_out;

  // workspace carve (f16 buffers), ~34.6 MB total
  char* w = (char*)d_ws;
  _Float16* x16   = (_Float16*)w; w += (size_t)ROWS * DM * 2;
  _Float16* wqkvT = (_Float16*)w; w += (size_t)NQKV * DM * 2;
  _Float16* woT   = (_Float16*)w; w += (size_t)DM * DM * 2;
  _Float16* qkv   = (_Float16*)w; w += (size_t)ROWS * NQKV * 2;
  _Float16* attn  = (_Float16*)w; w += (size_t)ROWS * (HH * HD) * 2;

  k_cvt_x<<<(ROWS * DM + 255) / 256, 256, 0, stream>>>(x, x16, ROWS * DM);
  k_pack_wqkv<<<(NQKV * DM + 255) / 256, 256, 0, stream>>>(Wq, Wk, Wv, wqkvT);
  k_pack_wo<<<(DM * DM + 255) / 256, 256, 0, stream>>>(Wo, woT);

  k_gemm<true><<<dim3(NQKV / 64, ROWS / 128), 256, 0, stream>>>(
      x16, wqkvT, (void*)qkv, ROWS, NQKV, DM);

  k_rope<<<((size_t)ROWS * (HH + HKV) * 32 + 255) / 256, 256, 0, stream>>>(qkv, pos);

  k_attn<<<dim3(SS / 64, BB * HH), 128, 0, stream>>>(qkv, attn);

  k_gemm<false><<<dim3(DM / 64, ROWS / 128), 256, 0, stream>>>(
      attn, woT, (void*)out, ROWS, DM, DM);
}
